// CNLP_5592047419469
// MI455X (gfx1250) — matrix operand order repaired and checked by reading, not compiled, not run on hardware
//
#include <hip/hip_runtime.h>
#include <hip/hip_bf16.h>

// Problem constants from the reference:
//   N, E, D_IN, H, OUT = (10000, 8192, 256, 256, 1)
#define NN 10000
#define KP 10016   // padded K-stride (elements) of transposed x; pad zero-filled
#define EE 8192
#define DD 256
#define BM 32      // edge-rows per block (two 16-row WMMA tiles per wave)

typedef __bf16 bf16_t;
typedef bf16_t v16bf __attribute__((ext_vector_type(16)));
typedef float  v8f   __attribute__((ext_vector_type(8)));
typedef float  f32x4 __attribute__((ext_vector_type(4)));
typedef unsigned short u16x8 __attribute__((ext_vector_type(8)));

__device__ __forceinline__ unsigned short f2b(float f) {
  union { bf16_t b; unsigned short u; } c;
  c.b = (bf16_t)f;
  return c.u;
}

__device__ __forceinline__ v16bf mkfrag(u16x8 lo, u16x8 hi) {
  union { u16x8 h[2]; v16bf v; } f;
  f.h[0] = lo; f.h[1] = hi;
  return f.v;
}

// ---- WMMA fragment builders (wave32, v_wmma_f32_16x16x32_bf16) -------------
// A (16x32 MxK): lanes 0-15 -> M=lane, K={0..7,16..23}; lanes 16-31 -> M=lane-16,
// K={8..15,24..31}; 2 consecutive K per VGPR, low K in [15:0].
__device__ __forceinline__ v16bf a_frag_lds(const unsigned short* As) {
  const int lane = threadIdx.x & 31;
  const unsigned short* p = As + (lane & 15) * 32 + ((lane < 16) ? 0 : 8);
  return mkfrag(*(const u16x8*)p, *(const u16x8*)(p + 16));
}
// A fragment directly from global row-major MxK bf16 (K multiple of 32).
__device__ __forceinline__ v16bf a_frag_glb(const unsigned short* A, int K,
                                            int row0, int k0) {
  const int lane = threadIdx.x & 31;
  const unsigned short* p =
      A + (size_t)(row0 + (lane & 15)) * K + k0 + ((lane < 16) ? 0 : 8);
  return mkfrag(*(const u16x8*)p, *(const u16x8*)(p + 16));
}
// B (32x16 KxN): column N = lane%16; lanes 0-15 hold K={0..7,16..23}, lanes 16-31 K={8..15,24..31}.
// Source layout [n][k] (K contiguous per column) with row stride `stride` elems.
__device__ __forceinline__ v16bf b_frag_glb(const unsigned short* B, int stride,
                                            int n0, int k0) {
  const int lane = threadIdx.x & 31;
  const unsigned short* p =
      B + (size_t)(n0 + (lane & 15)) * stride + k0 + ((lane < 16) ? 0 : 8);
  return mkfrag(*(const u16x8*)p, *(const u16x8*)(p + 16));
}

__device__ __forceinline__ v8f wmma_bf16(v16bf a, v16bf b, v8f c) {
  return __builtin_amdgcn_wmma_f32_16x16x32_bf16(false, a, false, b,
                                                 (short)0, c, false, false);
}

// ---------------- prep kernels ----------------------------------------------
__global__ __launch_bounds__(256) void cvt_rows_bf16(const float* __restrict__ src,
                                                     unsigned short* __restrict__ dst) {
  const size_t i = (size_t)blockIdx.x * DD + threadIdx.x;
  dst[i] = f2b(src[i]);
}

// Tiled transpose: xbT[d][k] = bf16(x[k][d]), zero-padded to KP columns.
__global__ __launch_bounds__(256) void transpose_x_bf16(const float* __restrict__ x,
                                                        unsigned short* __restrict__ xbT) {
  __shared__ unsigned short T[64][65];
  const int k0 = blockIdx.x * 64;
  const int d0 = blockIdx.y * 64;
  const int tid = threadIdx.x;
  #pragma unroll
  for (int it = 0; it < 16; ++it) {
    const int l = it * 256 + tid;
    const int kk = l >> 6, dd = l & 63;
    const int k = k0 + kk;
    const float v = (k < NN) ? x[(size_t)k * DD + d0 + dd] : 0.f;
    T[kk][dd] = f2b(v);
  }
  __syncthreads();
  #pragma unroll
  for (int it = 0; it < 16; ++it) {
    const int l = it * 256 + tid;
    const int dd = l >> 6, kk = l & 63;
    const int k = k0 + kk;
    if (k < KP) xbT[(size_t)(d0 + dd) * KP + k] = T[kk][dd];
  }
}

// p[e][d] = bf16(x[e0][d] * x[e1][d]). Grid: E blocks, 256 threads.
__global__ __launch_bounds__(256) void pair_prod_bf16(const float* __restrict__ x,
                                                      const long long* __restrict__ edge,
                                                      unsigned short* __restrict__ p) {
  const int e = blockIdx.x, d = threadIdx.x;
  const long long i = edge[(size_t)e * 2];
  const long long j = edge[(size_t)e * 2 + 1];
  p[(size_t)e * DD + d] = f2b(x[(size_t)i * DD + d] * x[(size_t)j * DD + d]);
}

// ---------------- dominant kernel: xcn = (adj[e0] .* adj[e1]) @ x -----------
// Block: 32 edges x 256 cols; 8 waves, each a 32x32 C tile (4 accumulators:
// 2 row-tiles sharing 2 B fragments -> 4 wmma per iteration). A tile (32x32
// products, exact 0/1 in bf16) double-buffered in LDS; B fragments stream from
// the L2-resident transposed bf16 x. One barrier per K-step of 32. adj is a
// 655 MB single-use stream -> non-temporal loads so it does not evict xbT.
__global__ __launch_bounds__(256) void xcn_wmma(const float* __restrict__ adj,
                                                const long long* __restrict__ edge,
                                                const unsigned short* __restrict__ xbT,
                                                unsigned short* __restrict__ xcnb) {
  __shared__ __align__(16) unsigned short As[2][BM * 32];
  __shared__ int sE0[BM], sE1[BM];

  const int tid = threadIdx.x;
  const int lane = tid & 31;
  const int wave = tid >> 5;
  const int rowBase = blockIdx.x * BM;

  if (tid < BM) {
    sE0[tid] = (int)edge[(size_t)(rowBase + tid) * 2];
    sE1[tid] = (int)edge[(size_t)(rowBase + tid) * 2 + 1];
  }
  __syncthreads();

  const int m  = tid >> 3;          // A-tile row this thread fills (0..31)
  const int kk = (tid & 7) * 4;     // four consecutive K per thread
  const float* r0 = adj + (size_t)sE0[m] * NN;
  const float* r1 = adj + (size_t)sE1[m] * NN;
  const int c0 = wave * 32;

  v8f acc00 = {}, acc01 = {}, acc10 = {}, acc11 = {};

  auto storeA = [&](unsigned short* dst, int k0) {
    const int k = k0 + kk;
    const f32x4 a = __builtin_nontemporal_load((const f32x4*)(r0 + k));
    const f32x4 b = __builtin_nontemporal_load((const f32x4*)(r1 + k));
    const unsigned long long v =
        (unsigned long long)f2b(a.x * b.x)        |
        ((unsigned long long)f2b(a.y * b.y) << 16) |
        ((unsigned long long)f2b(a.z * b.z) << 32) |
        ((unsigned long long)f2b(a.w * b.w) << 48);
    *(unsigned long long*)(dst + m * 32 + kk) = v;
  };
  auto storeA_masked = [&](unsigned short* dst, int k0) {
    const int k = k0 + kk;
    unsigned long long v = 0;
    #pragma unroll
    for (int i = 0; i < 4; ++i) {
      const float pi = (k + i < NN) ? r0[k + i] * r1[k + i] : 0.f;
      v |= (unsigned long long)f2b(pi) << (16 * i);
    }
    *(unsigned long long*)(dst + m * 32 + kk) = v;
  };
  auto compute = [&](const unsigned short* tile, int k0) {
    const v16bf a0 = a_frag_lds(tile);            // rows 0-15
    const v16bf a1 = a_frag_lds(tile + 16 * 32);  // rows 16-31
    const v16bf b0 = b_frag_glb(xbT, KP, c0, k0);
    const v16bf b1 = b_frag_glb(xbT, KP, c0 + 16, k0);
    acc00 = wmma_bf16(a0, b0, acc00);
    acc01 = wmma_bf16(a0, b1, acc01);
    acc10 = wmma_bf16(a1, b0, acc10);
    acc11 = wmma_bf16(a1, b1, acc11);
  };

  // Tiles t = 0..312 (k0 = 32t); tile 312 is the masked tail (K 9984..9999).
  storeA(As[0], 0);
  int buf = 0;
  for (int t = 0; t < 311; ++t) {
    __syncthreads();
    __builtin_prefetch((const void*)(r0 + 32 * t + 96), 0, 1);
    __builtin_prefetch((const void*)(r1 + 32 * t + 96), 0, 1);
    storeA(As[buf ^ 1], 32 * (t + 1));
    compute(As[buf], 32 * t);
    buf ^= 1;
  }
  __syncthreads();
  storeA_masked(As[buf ^ 1], 9984);   // tail tile
  compute(As[buf], 9952);
  buf ^= 1;
  __syncthreads();
  compute(As[buf], 9984);             // B pad columns are zero-filled

  // Epilogue: lanes 0-15: M=v, lanes 16-31: M=v+8; N = c0 + lane%16.
  const int n0 = c0 + (lane & 15);
  const int rh = (lane < 16) ? 0 : 8;
  #pragma unroll
  for (int v = 0; v < 8; ++v) {
    const size_t ra = (size_t)(rowBase + rh + v);
    const size_t rb = (size_t)(rowBase + 16 + rh + v);
    xcnb[ra * DD + n0]      = f2b(acc00[v]);
    xcnb[ra * DD + n0 + 16] = f2b(acc01[v]);
    xcnb[rb * DD + n0]      = f2b(acc10[v]);
    xcnb[rb * DD + n0 + 16] = f2b(acc11[v]);
  }
}

// ---------------- 256x256 dense layers: Y = act(A @ W^T + bias) -------------
// A: M x 256 bf16 row-major. Wb16: bf16 copy of W in ORIGINAL [h][d] layout --
// exactly the [n][k] layout the B fragment wants. 32x32 C per wave, no LDS,
// no barriers; K loop fully unrolled (32 wmma per wave).
template <bool RELU, bool OUTBF16>
__global__ __launch_bounds__(256) void gemm_direct(const unsigned short* __restrict__ A,
                                                   const unsigned short* __restrict__ Wb16,
                                                   const float* __restrict__ bias,
                                                   void* __restrict__ out) {
  const int tid = threadIdx.x;
  const int lane = tid & 31;
  const int wave = tid >> 5;
  const int rowBase = blockIdx.x * BM;
  const int c0 = wave * 32;

  v8f acc00 = {}, acc01 = {}, acc10 = {}, acc11 = {};
  #pragma unroll
  for (int k0 = 0; k0 < DD; k0 += 32) {
    const v16bf a0 = a_frag_glb(A, DD, rowBase, k0);
    const v16bf a1 = a_frag_glb(A, DD, rowBase + 16, k0);
    const v16bf b0 = b_frag_glb(Wb16, DD, c0, k0);
    const v16bf b1 = b_frag_glb(Wb16, DD, c0 + 16, k0);
    acc00 = wmma_bf16(a0, b0, acc00);
    acc01 = wmma_bf16(a0, b1, acc01);
    acc10 = wmma_bf16(a1, b0, acc10);
    acc11 = wmma_bf16(a1, b1, acc11);
  }

  const int n0 = c0 + (lane & 15);
  const int rh = (lane < 16) ? 0 : 8;
  #pragma unroll
  for (int v = 0; v < 8; ++v) {
    const size_t ra = (size_t)(rowBase + rh + v);
    const size_t rb = (size_t)(rowBase + 16 + rh + v);
    float f00 = acc00[v] + bias[n0];
    float f01 = acc01[v] + bias[n0 + 16];
    float f10 = acc10[v] + bias[n0];
    float f11 = acc11[v] + bias[n0 + 16];
    if (RELU) {
      f00 = fmaxf(f00, 0.f); f01 = fmaxf(f01, 0.f);
      f10 = fmaxf(f10, 0.f); f11 = fmaxf(f11, 0.f);
    }
    if (OUTBF16) {
      unsigned short* o = (unsigned short*)out;
      o[ra * DD + n0]      = f2b(f00);
      o[ra * DD + n0 + 16] = f2b(f01);
      o[rb * DD + n0]      = f2b(f10);
      o[rb * DD + n0 + 16] = f2b(f11);
    } else {
      float* o = (float*)out;
      o[ra * DD + n0]      = f00;
      o[ra * DD + n0 + 16] = f01;
      o[rb * DD + n0]      = f10;
      o[rb * DD + n0 + 16] = f11;
    }
  }
}

// ---------------- final: out[e] = bl + sum_h Wl[h]*(beta*h3 + xij) ----------
__global__ __launch_bounds__(256) void final_dot(const float* __restrict__ h3,
                                                 const float* __restrict__ xij,
                                                 const float* __restrict__ Wl,
                                                 const float* __restrict__ bl,
                                                 const float* __restrict__ beta,
                                                 float* __restrict__ out) {
  const int e = blockIdx.x * 8 + (threadIdx.x >> 5);
  const int lane = threadIdx.x & 31;
  const float bet = beta[0];
  float s = 0.f;
  #pragma unroll
  for (int i = lane; i < DD; i += 32) {
    const float a = __builtin_nontemporal_load(h3  + (size_t)e * DD + i);
    const float b = __builtin_nontemporal_load(xij + (size_t)e * DD + i);
    s += Wl[i] * (bet * a + b);
  }
  #pragma unroll
  for (int off = 16; off > 0; off >>= 1)
    s += __shfl_xor(s, off, 32);
  if (lane == 0) out[e] = s + bl[0];
}

// ---------------- host launcher ---------------------------------------------
extern "C" void kernel_launch(void* const* d_in, const int* in_sizes, int n_in,
                              void* d_out, int out_size, void* d_ws, size_t ws_size,
                              hipStream_t stream) {
  (void)in_sizes; (void)n_in; (void)out_size; (void)ws_size;
  const float*     x    = (const float*)d_in[0];
  const float*     adj  = (const float*)d_in[1];
  const long long* edge = (const long long*)d_in[2];  // reference uses int64
  const float* W1 = (const float*)d_in[3];  const float* b1 = (const float*)d_in[4];
  const float* W2 = (const float*)d_in[5];  const float* b2 = (const float*)d_in[6];
  const float* W3 = (const float*)d_in[7];  const float* b3 = (const float*)d_in[8];
  const float* Wa = (const float*)d_in[9];  const float* ba = (const float*)d_in[10];
  const float* Wb = (const float*)d_in[11]; const float* bb = (const float*)d_in[12];
  const float* Wl = (const float*)d_in[13]; const float* bl = (const float*)d_in[14];
  const float* beta = (const float*)d_in[15];
  float* out = (float*)d_out;

  char* ws = (char*)d_ws;
  size_t off = 0;
  auto alloc = [&](size_t bytes) -> void* {
    void* p = ws + off;
    off = (off + bytes + 255) & ~(size_t)255;
    return p;
  };
  unsigned short* xbT  = (unsigned short*)alloc((size_t)DD * KP * 2); // x^T bf16
  unsigned short* Wa16 = (unsigned short*)alloc((size_t)DD * DD * 2);
  unsigned short* Wb16 = (unsigned short*)alloc((size_t)DD * DD * 2);
  unsigned short* W116 = (unsigned short*)alloc((size_t)DD * DD * 2);
  unsigned short* W216 = (unsigned short*)alloc((size_t)DD * DD * 2);
  unsigned short* W316 = (unsigned short*)alloc((size_t)DD * DD * 2);
  unsigned short* p    = (unsigned short*)alloc((size_t)EE * DD * 2); // xi*xj bf16
  unsigned short* t1   = (unsigned short*)alloc((size_t)EE * DD * 2);
  float*          xij  = (float*)         alloc((size_t)EE * DD * 4);
  unsigned short* xcnb = (unsigned short*)alloc((size_t)EE * DD * 2);
  unsigned short* h1   = (unsigned short*)alloc((size_t)EE * DD * 2);
  unsigned short* h2   = (unsigned short*)alloc((size_t)EE * DD * 2);
  float*          h3   = (float*)         alloc((size_t)EE * DD * 4);

  // Prep: transposed bf16 x (L2-resident B operand), bf16 weight copies.
  transpose_x_bf16<<<dim3((KP + 63) / 64, DD / 64), 256, 0, stream>>>(x, xbT);
  cvt_rows_bf16<<<DD, 256, 0, stream>>>(Wa, Wa16);
  cvt_rows_bf16<<<DD, 256, 0, stream>>>(Wb, Wb16);
  cvt_rows_bf16<<<DD, 256, 0, stream>>>(W1, W116);
  cvt_rows_bf16<<<DD, 256, 0, stream>>>(W2, W216);
  cvt_rows_bf16<<<DD, 256, 0, stream>>>(W3, W316);

  // Edge-pair elementwise product (gather).
  pair_prod_bf16<<<EE, 256, 0, stream>>>(x, edge, p);

  // Dominant GEMM: xcn = (adj[e0] .* adj[e1]) @ x  (42 GFLOP, bf16 WMMA).
  xcn_wmma<<<EE / BM, 256, 0, stream>>>(adj, edge, xbT, xcnb);

  // xij = relu(p @ Wa^T + ba) @ Wb^T + bb
  gemm_direct<true,  true ><<<EE / BM, 256, 0, stream>>>(p,  Wa16, ba, t1);
  gemm_direct<false, false><<<EE / BM, 256, 0, stream>>>(t1, Wb16, bb, xij);

  // hcn MLP: relu -> relu -> linear
  gemm_direct<true,  true ><<<EE / BM, 256, 0, stream>>>(xcnb, W116, b1, h1);
  gemm_direct<true,  true ><<<EE / BM, 256, 0, stream>>>(h1,   W216, b2, h2);
  gemm_direct<false, false><<<EE / BM, 256, 0, stream>>>(h2,   W316, b3, h3);

  // out = (h3*beta + xij) @ Wl.T + bl
  final_dot<<<EE / 8, 256, 0, stream>>>(h3, xij, Wl, bl, beta, out);
}
